// c_RGLSTM_15272903704604
// MI455X (gfx1250) — matrix operand order repaired and checked by reading, not compiled, not run on hardware
//
#include <hip/hip_runtime.h>
#include <hip/hip_bf16.h>

typedef __attribute__((ext_vector_type(8)))  __bf16 bf16x8;
typedef __attribute__((ext_vector_type(16))) __bf16 v16bf;
typedef __attribute__((ext_vector_type(8)))  float  v8f;

constexpr int kH  = 512;   // hidden
constexpr int kB  = 64;    // batch
constexpr int kS  = 512;   // seq len
constexpr int kI  = 256;   // input
constexpr int kNB = 16;    // persistent workgroups

// ---------- helpers ----------
__device__ __forceinline__ __bf16 f2bf16(float f) {
  unsigned u = __builtin_bit_cast(unsigned, f);
  u += 0x7FFFu + ((u >> 16) & 1u);            // round-to-nearest-even
  return __builtin_bit_cast(__bf16, (unsigned short)(u >> 16));
}
__device__ __forceinline__ float sigmoid_f(float x) {
#if __has_builtin(__builtin_amdgcn_rcpf)
  return __builtin_amdgcn_rcpf(1.f + __expf(-x));
#else
  return 1.f / (1.f + __expf(-x));
#endif
}
__device__ __forceinline__ float tanh_f(float x) {
#if __has_builtin(__builtin_amdgcn_tanhf)
  return __builtin_amdgcn_tanhf(x);
#else
  return tanhf(x);
#endif
}
// fragment = two contiguous 8-element (16-byte) runs
__device__ __forceinline__ v16bf ldfrag(const __bf16* p0, const __bf16* p1) {
  bf16x8 lo = *(const bf16x8*)p0;
  bf16x8 hi = *(const bf16x8*)p1;
  return __builtin_shufflevector(lo, hi, 0,1,2,3,4,5,6,7,8,9,10,11,12,13,14,15);
}
// A frag (16x32): lo @ k+8*half, hi @ k+16+8*half
__device__ __forceinline__ v16bf ldA(const __bf16* row, int kk, int half) {
  return ldfrag(row + kk + 8 * half, row + kk + 16 + 8 * half);
}
// B frag (32x16): lo @ k+8*half, hi @ k+16+8*half
__device__ __forceinline__ v16bf ldB(const __bf16* row, int kk, int half) {
  return ldfrag(row + kk + 8 * half, row + kk + 16 + 8 * half);
}
// tiled pre-activation layout: [t][nt(128)][mt(4)][half(2)][lane(16)][r(8)]
__device__ __forceinline__ size_t preIdx(int t, int nt, int mt, int half, int lm) {
  return (((((size_t)t * 128 + nt) * 4 + mt) * 2 + half) * 16 + lm) * 8;
}

__device__ __forceinline__ void gridBarrier(unsigned* cnt, unsigned target) {
  __threadfence();
  __syncthreads();
  if (threadIdx.x == 0) {
    atomicAdd(cnt, 1u);
    while (__hip_atomic_load(cnt, __ATOMIC_RELAXED, __HIP_MEMORY_SCOPE_AGENT) < target)
      __builtin_amdgcn_s_sleep(1);
  }
  __syncthreads();
  __threadfence();
}

// =====================================================================
// f32 -> bf16 bulk converter (8 elements/thread, vector loads/stores)
// =====================================================================
__global__ void __launch_bounds__(256)
cvt_bf16(const float* __restrict__ src, __bf16* __restrict__ dst, long n) {
  long i = ((long)blockIdx.x * 256 + threadIdx.x) * 8;
  if (i >= n) return;
  bf16x8 v;
#pragma unroll
  for (int e = 0; e < 8; ++e) v[e] = f2bf16(src[i + e]);
  *(bf16x8*)(dst + i) = v;
}

// =====================================================================
// Phase A: pre[t][b][:] = x[b,t,:] @ W_ih^T + b_ih + b_hh  (parallel)
// grid = (2048/128, S), block = 256; output in WMMA C-fragment tiling.
// =====================================================================
__global__ void __launch_bounds__(256)
preact_gemm(const __bf16* __restrict__ xbf, const __bf16* __restrict__ Wbf,
            const float* __restrict__ bih, const float* __restrict__ bhh,
            float* __restrict__ pre) {
  const int lane = threadIdx.x & 31, wave = threadIdx.x >> 5;
  const int half = lane >> 4, lm = lane & 15;
  const int mt = wave & 3, ng = wave >> 2;            // ng in 0..1
  const int t = blockIdx.y;                           // row block == timestep
  const int nbase0 = blockIdx.x * 128 + ng * 64;

  v8f acc[4];
#pragma unroll
  for (int j = 0; j < 4; ++j)
#pragma unroll
    for (int r = 0; r < 8; ++r) acc[j][r] = 0.f;

  const int m = mt * 16 + lm;                         // batch row
  const __bf16* xrow = xbf + ((size_t)m * kS + t) * kI;   // x is (B,S,I)

  for (int kk = 0; kk < kI; kk += 32) {
    const v16bf a = ldA(xrow, kk, half);
#pragma unroll
    for (int j = 0; j < 4; ++j) {
      const int n = nbase0 + j * 16 + lm;
      const v16bf bfrag = ldB(Wbf + (size_t)n * kI, kk, half);
      acc[j] = __builtin_amdgcn_wmma_f32_16x16x32_bf16(
          false, a, false, bfrag, (short)0, acc[j], false, false);
    }
  }
#pragma unroll
  for (int j = 0; j < 4; ++j) {
    const int n = nbase0 + j * 16 + lm;
    const int nt = (nbase0 >> 4) + j;
    const float bias = bih[n] + bhh[n];
    v8f outv;
#pragma unroll
    for (int r = 0; r < 8; ++r) outv[r] = acc[j][r] + bias;
    *(v8f*)(pre + preIdx(t, nt, mt, half, lm)) = outv;   // coalesced 32B store
  }
}

// =====================================================================
// Phase B: persistent scan. 16 WGs x 256 threads; WG w owns hidden cols
// J=[32w,32w+32). LDS: W_hh slice [128][512] + G slice transposed [64][512]
// (192 KB). One grid barrier/step; ping-pong bf16 h/c broadcast buffers.
// Software pipeline: next step's pre-activation tiles are loaded during
// the stats GEMM so the post-barrier gates GEMM starts immediately.
// =====================================================================
__global__ void __launch_bounds__(256)
rglstm_scan(const float* __restrict__ pre, const float* __restrict__ eps,
            const float* __restrict__ Whh, const float* __restrict__ G,
            __bf16* __restrict__ hbuf, __bf16* __restrict__ cbuf,
            float* __restrict__ hid, float* __restrict__ cmu,
            float* __restrict__ cstd, float* __restrict__ hf,
            float* __restrict__ cf, unsigned* __restrict__ cnt) {
  extern __shared__ __bf16 smem[];
  __bf16* sW  = smem;               // [128][512] row-major (gate rows)
  __bf16* sGt = smem + 128 * 512;   // [64][512]  col-major slice of G

  const int wg = blockIdx.x;
  const int tid = threadIdx.x;
  const int lane = tid & 31, wave = tid >> 5;
  const int half = lane >> 4, lm = lane & 15;
  const int mt = wave & 3, jt = wave >> 2;   // jt in 0..1
  const int J0 = wg * 32;
  const int jg = J0 + jt * 16 + lm;          // this lane's hidden column
  const int mrow = mt * 16 + lm;             // this lane's A-matrix row

  // --- stage resident weight slices (f32 -> bf16) ---
  for (int idx = tid; idx < 128 * 512; idx += 256) {      // k fast -> coalesced
    int rl = idx >> 9, k = idx & 511;
    int g = rl >> 5, jj = rl & 31;
    sW[idx] = f2bf16(Whh[(size_t)(g * kH + J0 + jj) * kH + k]);
  }
  for (int idx = tid; idx < 512 * 64; idx += 256) {       // c fast -> coalesced
    int k = idx >> 6, c = idx & 63;
    int col = (c < 32) ? (J0 + c) : (kH + J0 + (c - 32));
    sGt[(size_t)c * 512 + k] = f2bf16(G[(size_t)k * (2 * kH) + col]);
  }
  __syncthreads();

  // --- t=0: h0=0, c0=0 => stats=0 => mu0=clip(0)=1e-6, std0=softplus(0)=ln2
  const float mu0 = 1e-6f, std0 = 0.69314718055994531f;
  const __bf16 z = f2bf16(0.f);
  float cloc[8];
#pragma unroll
  for (int r = 0; r < 8; ++r) {
    const int m = mt * 16 + r + 8 * half;
    hbuf[(size_t)m * kH + jg] = z;                        // parity 0
    const float e0 = eps[(size_t)m * kH + jg];            // eps[0][m][jg]
    cloc[r] = mu0 + e0 * std0;
    cmu [((size_t)m * (kS + 1)) * kH + jg] = mu0;
    cstd[((size_t)m * (kS + 1)) * kH + jg] = std0;
  }

  // prime the pre-activation accumulators for t=0
  v8f acc[4];
#pragma unroll
  for (int g = 0; g < 4; ++g)
    acc[g] = *(const v8f*)(pre + preIdx(0, g * 32 + wg * 2 + jt, mt, half, lm));

  unsigned bar = 0;
  gridBarrier(cnt, ++bar * kNB);

  for (int t = 0; t < kS; ++t) {
    const __bf16* hcur = hbuf + (size_t)(t & 1) * (kB * kH);
    __bf16* hnext  = hbuf + (size_t)((t + 1) & 1) * (kB * kH);
    __bf16* cshare = cbuf + (size_t)(t & 1) * (kB * kH);

    // prefetch eps[t+1] for the stats phase (one GEMM ahead of use)
    float ev[8];
#pragma unroll
    for (int r = 0; r < 8; ++r) {
      const int m = mt * 16 + r + 8 * half;
      ev[r] = eps[((size_t)(t + 1) * kB + m) * kH + jg];
    }

    // ---- gates = pre[t] + h @ W_hh^T (4 gate tiles per wave) ----
    const __bf16* harow = hcur + (size_t)mrow * kH;
    for (int kk = 0; kk < kH; kk += 32) {
      const v16bf a = ldA(harow, kk, half);
#pragma unroll
      for (int g = 0; g < 4; ++g) {
        const int rl = g * 32 + jt * 16 + lm;             // LDS W row
        const v16bf bfrag = ldB(sW + (size_t)rl * kH, kk, half);
        acc[g] = __builtin_amdgcn_wmma_f32_16x16x32_bf16(
            false, a, false, bfrag, (short)0, acc[g], false, false);
      }
    }
    // ---- LSTM elementwise (positions align with accumulator layout) ----
#pragma unroll
    for (int r = 0; r < 8; ++r) {
      const int m = mt * 16 + r + 8 * half;
      const float ig = sigmoid_f(acc[0][r]);
      const float fg = sigmoid_f(acc[1][r]);
      const float gg = tanh_f(acc[2][r]);
      const float og = sigmoid_f(acc[3][r]);
      const float cn = fg * cloc[r] + ig * gg;
      const float hn = og * tanh_f(cn);
      hnext [(size_t)m * kH + jg] = f2bf16(hn);
      cshare[(size_t)m * kH + jg] = f2bf16(cn);
      hid[((size_t)m * kS + t) * kH + jg] = hn;
      if (t == kS - 1) hf[(size_t)m * kH + jg] = hn;
    }
    gridBarrier(cnt, ++bar * kNB);   // publish h_new + c_new to all WGs

    // software pipeline: load next step's pre tiles; they complete while
    // the stats GEMM below executes (no cross-WG dependency).
    if (t + 1 < kS) {
#pragma unroll
      for (int g = 0; g < 4; ++g)
        acc[g] = *(const v8f*)(pre +
                 preIdx(t + 1, g * 32 + wg * 2 + jt, mt, half, lm));
    }

    // ---- stats = c_new @ G : one mu tile + one std tile per wave ----
    v8f smu, sst;
#pragma unroll
    for (int r = 0; r < 8; ++r) { smu[r] = 0.f; sst[r] = 0.f; }
    const __bf16* carow = cshare + (size_t)mrow * kH;
    const __bf16* gmu = sGt + (size_t)(jt * 16 + lm) * 512;        // mu col
    const __bf16* gsd = sGt + (size_t)(32 + jt * 16 + lm) * 512;   // std col
    for (int kk = 0; kk < kH; kk += 32) {
      const v16bf a = ldA(carow, kk, half);
      const v16bf bmu = ldB(gmu, kk, half);
      const v16bf bsd = ldB(gsd, kk, half);
      smu = __builtin_amdgcn_wmma_f32_16x16x32_bf16(
          false, a, false, bmu, (short)0, smu, false, false);
      sst = __builtin_amdgcn_wmma_f32_16x16x32_bf16(
          false, a, false, bsd, (short)0, sst, false, false);
    }
#pragma unroll
    for (int r = 0; r < 8; ++r) {
      const int m = mt * 16 + r + 8 * half;
      const float mu = fminf(fmaxf(smu[r], 1e-6f), 1e6f);
      const float sv = sst[r];
      const float sp = fmaxf(sv, 0.f) + log1pf(__expf(-fabsf(sv)));
      const float sd = fmaxf(sp, 1e-6f);
      cloc[r] = mu + ev[r] * sd;                          // sampled cell (local)
      cmu [((size_t)m * (kS + 1) + (t + 1)) * kH + jg] = mu;
      cstd[((size_t)m * (kS + 1) + (t + 1)) * kH + jg] = sd;
      if (t == kS - 1) cf[(size_t)m * kH + jg] = cloc[r];
    }
  }
}

extern "C" void kernel_launch(void* const* d_in, const int* in_sizes, int n_in,
                              void* d_out, int out_size, void* d_ws, size_t ws_size,
                              hipStream_t stream) {
  const float* x   = (const float*)d_in[0];
  const float* eps = (const float*)d_in[1];
  const float* Wih = (const float*)d_in[2];
  const float* Whh = (const float*)d_in[3];
  const float* bih = (const float*)d_in[4];
  const float* bhh = (const float*)d_in[5];
  const float* G   = (const float*)d_in[6];

  float* out  = (float*)d_out;
  float* hid  = out;                                   // (B,S,H)
  float* cmu  = hid  + (size_t)kB * kS * kH;           // (B,S+1,H)
  float* cstd = cmu  + (size_t)kB * (kS + 1) * kH;     // (B,S+1,H)
  float* hf   = cstd + (size_t)kB * (kS + 1) * kH;     // (B,H)
  float* cf   = hf   + (size_t)kB * kH;                // (B,H)

  char* ws = (char*)d_ws;
  unsigned* cnt = (unsigned*)ws;                                    // 1 KB
  __bf16* hbuf = (__bf16*)(ws + 1024);                              // 2*B*H
  __bf16* cbuf = (__bf16*)(ws + 1024 + 2 * kB * kH * 2);            // 2*B*H
  float*  pre  = (float*) (ws + 1024 + 4 * kB * kH * 2);            // S*B*4H f32
  const size_t preBytes = (size_t)kS * kB * 4 * kH * sizeof(float); // 256 MB
  __bf16* xbf  = (__bf16*)((char*)pre + preBytes);                  // B*S*I
  __bf16* Wbf  = (__bf16*)((char*)xbf + (size_t)kB * kS * kI * 2);  // 4H*I

  hipMemsetAsync(cnt, 0, 1024, stream);

  const long nx = (long)kB * kS * kI;        // 8,388,608
  const long nw = (long)4 * kH * kI;         // 524,288
  cvt_bf16<<<(unsigned)((nx / 8 + 255) / 256), 256, 0, stream>>>(x,   xbf, nx);
  cvt_bf16<<<(unsigned)((nw / 8 + 255) / 256), 256, 0, stream>>>(Wih, Wbf, nw);

  dim3 gA(4 * kH / 128, kS);                 // (16, 512)
  preact_gemm<<<gA, 256, 0, stream>>>(xbf, Wbf, bih, bhh, pre);

  const size_t ldsBytes = (128 * 512 + 64 * 512) * sizeof(__bf16);  // 192 KB
  rglstm_scan<<<kNB, 256, ldsBytes, stream>>>(pre, eps, Whh, G, hbuf, cbuf,
                                              hid, cmu, cstd, hf, cf, cnt);
}
